// MAMBA_90469191123251
// MI455X (gfx1250) — compile-verified, matrix operand order checked by reading, not run on hardware
//
#include <hip/hip_runtime.h>
#include <hip/hip_fp16.h>

typedef __attribute__((ext_vector_type(16))) _Float16 v16h;
typedef __attribute__((ext_vector_type(8)))  _Float16 v8h;
typedef __attribute__((ext_vector_type(8)))  float    v8f;

#define DIMC    96
#define DINNER  192
#define NB      4
#define LSEQ    4096
#define MROWS   (NB*LSEQ)   // 16384

// ---------------------------------------------------------------------------
// fragment loader: 16 contiguous-pair f16 (two aligned 16B loads)
// A/B 16x32 f16 layout: element j <-> K = kb + (j<8 ? j : j+8),
// kb = kstep + (lane>>4)*8, row/col = lane&15
// ---------------------------------------------------------------------------
__device__ __forceinline__ v16h load_frag_h(const _Float16* __restrict__ row, int kb) {
    v8h a0 = *(const v8h*)(row + kb);
    v8h a1 = *(const v8h*)(row + kb + 16);
    v16h f;
    #pragma unroll
    for (int i = 0; i < 8; i++) { f[i] = a0[i]; f[i + 8] = a1[i]; }
    return f;
}

// ---------------------------------------------------------------------------
// 0) zero small stats buffer (graph-safe init each call)
// ---------------------------------------------------------------------------
__global__ void k_zero(float* p, int n) {
    int i = blockIdx.x * blockDim.x + threadIdx.x;
    if (i < n) p[i] = 0.f;
}

// 0b) f32 -> f16 weight pre-cast (done once per call; weights stay hot in L2)
__global__ void k_cast_f16(const float* __restrict__ src, _Float16* __restrict__ dst, int n) {
    int i = blockIdx.x * blockDim.x + threadIdx.x;
    if (i < n) dst[i] = (_Float16)src[i];
}

// ---------------------------------------------------------------------------
// 1) grouped 3x3 conv (96 groups x 3 in-ch) + per-channel sum/sumsq for BN
// ---------------------------------------------------------------------------
__global__ void k_conv3_bnstats(const float* __restrict__ x1,
                                const float* __restrict__ x2,
                                const float* __restrict__ cw,
                                const float* __restrict__ cb,
                                float* __restrict__ yraw,   // (MROWS,96)
                                float* __restrict__ stats)  // [96]=sum [96..191]=sumsq
{
    int b = blockIdx.x / DIMC;
    int c = blockIdx.x % DIMC;
    __shared__ float w[27];
    __shared__ float rs[256], rq[256];
    if (threadIdx.x < 27) w[threadIdx.x] = cw[c * 27 + threadIdx.x];
    __syncthreads();

    const float* in;
    if (c < 64) in = x1 + ((size_t)b * 192 + 3 * c) * (size_t)LSEQ;
    else        in = x2 + ((size_t)b * 96 + (3 * c - 192)) * (size_t)LSEQ;
    float bias = cb[c];

    float s = 0.f, sq = 0.f;
    for (int hw = threadIdx.x; hw < LSEQ; hw += blockDim.x) {
        int h = hw >> 6, wx = hw & 63;
        float acc = bias;
        #pragma unroll
        for (int ic = 0; ic < 3; ic++) {
            const float* p = in + (size_t)ic * LSEQ;
            #pragma unroll
            for (int ky = 0; ky < 3; ky++) {
                int yy = h + ky - 1;
                if (yy < 0 || yy > 63) continue;
                #pragma unroll
                for (int kx = 0; kx < 3; kx++) {
                    int xx = wx + kx - 1;
                    if (xx < 0 || xx > 63) continue;
                    acc += w[ic * 9 + ky * 3 + kx] * p[yy * 64 + xx];
                }
            }
        }
        yraw[((size_t)(b * LSEQ + hw)) * DIMC + c] = acc;
        s += acc; sq += acc * acc;
    }
    rs[threadIdx.x] = s; rq[threadIdx.x] = sq;
    __syncthreads();
    for (int st = 128; st > 0; st >>= 1) {
        if (threadIdx.x < st) { rs[threadIdx.x] += rs[threadIdx.x + st];
                                rq[threadIdx.x] += rq[threadIdx.x + st]; }
        __syncthreads();
    }
    if (threadIdx.x == 0) {
        atomicAdd(&stats[c],      rs[0]);
        atomicAdd(&stats[96 + c], rq[0]);
    }
}

// ---------------------------------------------------------------------------
// 2) BN apply, in place on (MROWS,96) buffer -> res
// ---------------------------------------------------------------------------
__global__ void k_bn_apply(float* __restrict__ y,
                           const float* __restrict__ stats,
                           const float* __restrict__ g,
                           const float* __restrict__ bb)
{
    int idx = blockIdx.x * blockDim.x + threadIdx.x;   // MROWS*96
    int c = idx % DIMC;
    const float invN = 1.f / (float)MROWS;
    float mu  = stats[c] * invN;
    float var = stats[96 + c] * invN - mu * mu;
    float sc  = g[c] * rsqrtf(var + 1e-5f);
    y[idx] = (y[idx] - mu) * sc + bb[c];
}

// ---------------------------------------------------------------------------
// 3) LayerNorm of u = x2 reshaped (b,l,96) -> f16 rows for WMMA
// ---------------------------------------------------------------------------
__global__ void k_layernorm(const float* __restrict__ x2,
                            const float* __restrict__ lg,
                            const float* __restrict__ lb,
                            _Float16* __restrict__ unh)   // (MROWS,96) f16
{
    int m = blockIdx.x;
    int b = m >> 12, l = m & (LSEQ - 1);
    int t = threadIdx.x;
    __shared__ float red[128];
    float u = 0.f;
    if (t < DIMC) u = x2[((size_t)b * DIMC + t) * (size_t)LSEQ + l];
    red[t] = (t < DIMC) ? u : 0.f;
    __syncthreads();
    for (int s = 64; s > 0; s >>= 1) { if (t < s) red[t] += red[t + s]; __syncthreads(); }
    float mu = red[0] * (1.f / 96.f);
    __syncthreads();
    float du = u - mu;
    red[t] = (t < DIMC) ? du * du : 0.f;
    __syncthreads();
    for (int s = 64; s > 0; s >>= 1) { if (t < s) red[t] += red[t + s]; __syncthreads(); }
    float var = red[0] * (1.f / 96.f);
    if (t < DIMC)
        unh[(size_t)m * DIMC + t] = (_Float16)(du * rsqrtf(var + 1e-6f) * lg[t] + lb[t]);
}

// ---------------------------------------------------------------------------
// 4) WMMA GEMM: xz(M,384) = un(M,96) @ in_proj_w(384,96)^T
//    wave computes 64(M)x16(N): B fragment reused across 4 WMMAs per k-step
// ---------------------------------------------------------------------------
__global__ __launch_bounds__(256)
void k_gemm_inproj(const _Float16* __restrict__ A,   // MROWS x 96 (f16)
                   const _Float16* __restrict__ Wh,  // 384 x 96 (f16)
                   float* __restrict__ Cout)         // MROWS x 384
{
    const int NT = 384 / 16;                         // 24 n-tiles
    int wave = (blockIdx.x << 3) + (threadIdx.x >> 5);
    int mg = wave / NT, nt = wave % NT;              // mg: 64-row group [0,256)
    int lane = threadIdx.x & 31;
    int r = lane & 15, hg = lane >> 4;

    const _Float16* arow = A  + (size_t)(mg * 64 + r) * DIMC;
    const _Float16* brow = Wh + (size_t)(nt * 16 + r) * DIMC;

    v8f acc[4] = {};
    #pragma unroll
    for (int kk = 0; kk < 96; kk += 32) {
        int kb = kk + hg * 8;
        v16h bf = load_frag_h(brow, kb);
        #pragma unroll
        for (int t = 0; t < 4; t++) {
            v16h af = load_frag_h(arow + (size_t)t * 16 * DIMC, kb);
            acc[t] = __builtin_amdgcn_wmma_f32_16x16x32_f16(false, af, false, bf,
                                                            (short)0, acc[t], false, false);
        }
    }
    #pragma unroll
    for (int t = 0; t < 4; t++) {
        float* crow = Cout + (size_t)(mg * 64 + t * 16 + hg * 8) * 384 + nt * 16 + r;
        #pragma unroll
        for (int i = 0; i < 8; i++) crow[(size_t)i * 384] = acc[t][i];
    }
}

// ---------------------------------------------------------------------------
// 5) causal depthwise conv1d (window 4, left-pad 3) + SiLU
// ---------------------------------------------------------------------------
__global__ void k_conv1d_silu(const float* __restrict__ xz,   // (MROWS,384)
                              const float* __restrict__ w,    // (192,4)
                              const float* __restrict__ bias, // (192)
                              float* __restrict__ xs)         // (MROWS,192)
{
    int idx = blockIdx.x * blockDim.x + threadIdx.x;  // MROWS*192
    int d = idx % DINNER;
    int m = idx / DINNER;
    int l = m & (LSEQ - 1), b = m >> 12;
    float acc = bias[d];
    #pragma unroll
    for (int j = 0; j < 4; j++) {
        int ls = l + j - 3;
        if (ls >= 0)
            acc += w[d * 4 + j] * xz[((size_t)(b * LSEQ + ls)) * 384 + d];
    }
    xs[(size_t)idx] = acc / (1.f + __expf(-acc));     // silu
}

// ---------------------------------------------------------------------------
// 6) x_proj (N=38) + dt_proj (K=6) + softplus; one block per row
// ---------------------------------------------------------------------------
__global__ void k_xproj_dt(const float* __restrict__ xs,      // (MROWS,192)
                           const float* __restrict__ xpw,     // (38,192)
                           const float* __restrict__ dtw,     // (192,6)
                           const float* __restrict__ dtbias,  // (192)
                           float* __restrict__ dtout,         // (MROWS,192)
                           float* __restrict__ bc)            // (MROWS,32)  B|C
{
    int m = blockIdx.x;
    int t = threadIdx.x;                                      // 64 threads
    __shared__ float xr[DINNER];
    __shared__ float dbl[38];
    for (int i = t; i < DINNER; i += 64) xr[i] = xs[(size_t)m * DINNER + i];
    __syncthreads();
    if (t < 38) {
        float a = 0.f;
        const float* wr = xpw + t * DINNER;
        for (int k = 0; k < DINNER; k++) a += wr[k] * xr[k];
        dbl[t] = a;
    }
    __syncthreads();
    if (t < 32) bc[(size_t)m * 32 + t] = dbl[6 + t];
    for (int d = t; d < DINNER; d += 64) {
        float a = dtbias[d];
        #pragma unroll
        for (int r = 0; r < 6; r++) a += dtw[d * 6 + r] * dbl[r];
        dtout[(size_t)m * DINNER + d] = (a > 20.f) ? a : log1pf(__expf(a)); // softplus
    }
}

// ---------------------------------------------------------------------------
// 7) selective scan over L; block per batch, thread per channel d (192),
//    16-float state in registers, B/C staged in LDS per step, streams
//    prefetched 8 steps ahead (global_prefetch_b8).
// ---------------------------------------------------------------------------
__global__ __launch_bounds__(192)
void k_scan(const float* __restrict__ dtb,   // (MROWS,192)
            const float* __restrict__ bc,    // (MROWS,32)
            const float* __restrict__ xs,    // (MROWS,192)
            const float* __restrict__ xz,    // (MROWS,384) z at +192
            const float* __restrict__ A_log, // (192,16)
            const float* __restrict__ Dp,    // (192)
            _Float16* __restrict__ G)        // (MROWS,192) f16
{
    int b = blockIdx.x;
    int d = threadIdx.x;
    __shared__ float Bs[16], Cs[16];
    float An[16], h[16];
    #pragma unroll
    for (int n = 0; n < 16; n++) { An[n] = -__expf(A_log[d * 16 + n]); h[n] = 0.f; }
    float Dd = Dp[d];
    for (int l = 0; l < LSEQ; l++) {
        int m = b * LSEQ + l;
        if (l + 8 < LSEQ) {                    // stream-ahead prefetch
            int mp = m + 8;
            __builtin_prefetch(dtb + (size_t)mp * DINNER + d, 0, 0);
            __builtin_prefetch(xs  + (size_t)mp * DINNER + d, 0, 0);
            __builtin_prefetch(xz  + (size_t)mp * 384 + DINNER + d, 0, 0);
        }
        if (d < 32) {
            float v = bc[(size_t)m * 32 + d];
            if (d < 16) Bs[d] = v; else Cs[d - 16] = v;
        }
        __syncthreads();
        float dt = dtb[(size_t)m * DINNER + d];
        float xv = xs[(size_t)m * DINNER + d];
        float y = 0.f;
        #pragma unroll
        for (int n = 0; n < 16; n++) {
            float dA = __expf(dt * An[n]);
            h[n] = dA * h[n] + dt * Bs[n] * xv;
            y += h[n] * Cs[n];
        }
        float zv = xz[(size_t)m * 384 + DINNER + d];
        float g  = (y + xv * Dd) * (zv / (1.f + __expf(-zv)));
        G[(size_t)m * DINNER + d] = (_Float16)g;
        __syncthreads();
    }
}

// ---------------------------------------------------------------------------
// 8) WMMA GEMM: out(M,96) = G(M,192) @ out_proj_w(96,192)^T + res
//    wave computes 64(M)x16(N); K=192 -> 6 k-steps x 4 WMMAs
// ---------------------------------------------------------------------------
__global__ __launch_bounds__(256)
void k_gemm_outproj(const _Float16* __restrict__ A,   // MROWS x 192 (f16)
                    const _Float16* __restrict__ Wh,  // 96 x 192 (f16)
                    const float* __restrict__ res,    // MROWS x 96
                    float* __restrict__ Cout)         // MROWS x 96
{
    const int NT = 96 / 16;                           // 6 n-tiles
    int wave = (blockIdx.x << 3) + (threadIdx.x >> 5);
    int mg = wave / NT, nt = wave % NT;
    int lane = threadIdx.x & 31;
    int r = lane & 15, hg = lane >> 4;

    const _Float16* arow = A  + (size_t)(mg * 64 + r) * DINNER;
    const _Float16* brow = Wh + (size_t)(nt * 16 + r) * DINNER;

    v8f acc[4] = {};
    #pragma unroll
    for (int kk = 0; kk < 192; kk += 32) {
        int kb = kk + hg * 8;
        v16h bf = load_frag_h(brow, kb);
        #pragma unroll
        for (int t = 0; t < 4; t++) {
            v16h af = load_frag_h(arow + (size_t)t * 16 * DINNER, kb);
            acc[t] = __builtin_amdgcn_wmma_f32_16x16x32_f16(false, af, false, bf,
                                                            (short)0, acc[t], false, false);
        }
    }
    #pragma unroll
    for (int t = 0; t < 4; t++) {
        size_t base = (size_t)(mg * 64 + t * 16 + hg * 8) * DIMC + nt * 16 + r;
        #pragma unroll
        for (int i = 0; i < 8; i++)
            Cout[base + (size_t)i * DIMC] = acc[t][i] + res[base + (size_t)i * DIMC];
    }
}

// ---------------------------------------------------------------------------
// 9) final 1x1 grouped down conv: 48 groups of 2 channels
// ---------------------------------------------------------------------------
__global__ void k_down(const float* __restrict__ outp,  // (MROWS,96)
                       const float* __restrict__ dw,    // (48,2)
                       float* __restrict__ dout)        // (NB,48,64,64)
{
    int idx = blockIdx.x * blockDim.x + threadIdx.x;    // NB*48*4096
    int hw = idx & (LSEQ - 1);
    int t  = idx >> 12;
    int oc = t % 48, b = t / 48;
    const float* row = outp + ((size_t)(b * LSEQ + hw)) * DIMC + 2 * oc;
    dout[idx] = dw[oc * 2] * row[0] + dw[oc * 2 + 1] * row[1];
}

// ---------------------------------------------------------------------------
extern "C" void kernel_launch(void* const* d_in, const int* in_sizes, int n_in,
                              void* d_out, int out_size, void* d_ws, size_t ws_size,
                              hipStream_t stream)
{
    const float* x1       = (const float*)d_in[0];
    const float* x2       = (const float*)d_in[1];
    const float* conv_w   = (const float*)d_in[2];
    const float* conv_b   = (const float*)d_in[3];
    const float* bn_g     = (const float*)d_in[4];
    const float* bn_b     = (const float*)d_in[5];
    const float* ln_g     = (const float*)d_in[6];
    const float* ln_b     = (const float*)d_in[7];
    const float* in_pw    = (const float*)d_in[8];
    const float* c1d_w    = (const float*)d_in[9];
    const float* c1d_b    = (const float*)d_in[10];
    const float* x_pw     = (const float*)d_in[11];
    const float* dt_pw    = (const float*)d_in[12];
    const float* dt_pb    = (const float*)d_in[13];
    const float* A_log    = (const float*)d_in[14];
    const float* Dp       = (const float*)d_in[15];
    const float* out_pw   = (const float*)d_in[16];
    const float* down_w   = (const float*)d_in[17];
    float* dout = (float*)d_out;

    char* ws = (char*)d_ws;
    size_t off = 0;
    auto alloc = [&](size_t bytes) -> void* {
        void* p = ws + off;
        off = (off + bytes + 255) & ~(size_t)255;
        return p;
    };
    float*    resbuf = (float*)   alloc((size_t)MROWS * DIMC   * 4);
    _Float16* unh    = (_Float16*)alloc((size_t)MROWS * DIMC   * 2);
    float*    xz     = (float*)   alloc((size_t)MROWS * 384    * 4);
    float*    xsb    = (float*)   alloc((size_t)MROWS * DINNER * 4);
    float*    dtb    = (float*)   alloc((size_t)MROWS * DINNER * 4);
    float*    bcb    = (float*)   alloc((size_t)MROWS * 32     * 4);
    _Float16* Gh     = (_Float16*)alloc((size_t)MROWS * DINNER * 2);
    float*    outp   = (float*)   alloc((size_t)MROWS * DIMC   * 4);
    _Float16* inpwh  = (_Float16*)alloc((size_t)384 * DIMC * 2);
    _Float16* outpwh = (_Float16*)alloc((size_t)DIMC * DINNER * 2);
    float*    stats  = (float*)   alloc(192 * 4);

    k_zero<<<1, 192, 0, stream>>>(stats, 192);
    k_cast_f16<<<(384 * DIMC + 255) / 256, 256, 0, stream>>>(in_pw, inpwh, 384 * DIMC);
    k_cast_f16<<<(DIMC * DINNER + 255) / 256, 256, 0, stream>>>(out_pw, outpwh, DIMC * DINNER);
    k_conv3_bnstats<<<NB * DIMC, 256, 0, stream>>>(x1, x2, conv_w, conv_b, resbuf, stats);
    k_bn_apply<<<(MROWS * DIMC) / 256, 256, 0, stream>>>(resbuf, stats, bn_g, bn_b);
    k_layernorm<<<MROWS, 128, 0, stream>>>(x2, ln_g, ln_b, unh);
    k_gemm_inproj<<<(MROWS / 64) * (384 / 16) / 8, 256, 0, stream>>>(unh, inpwh, xz);
    k_conv1d_silu<<<(MROWS * DINNER) / 256, 256, 0, stream>>>(xz, c1d_w, c1d_b, xsb);
    k_xproj_dt<<<MROWS, 64, 0, stream>>>(xsb, x_pw, dt_pw, dt_pb, dtb, bcb);
    k_scan<<<NB, DINNER, 0, stream>>>(dtb, bcb, xsb, xz, A_log, Dp, Gh);
    k_gemm_outproj<<<(MROWS / 64) * (DIMC / 16) / 8, 256, 0, stream>>>(Gh, outpwh, resbuf, outp);
    k_down<<<(NB * 48 * LSEQ) / 256, 256, 0, stream>>>(outp, down_w, dout);
}
